// BMN_481036337693
// MI455X (gfx1250) — compile-verified, matrix operand order checked by reading, not run on hardware
//
#include <hip/hip_runtime.h>
#include <hip/hip_bf16.h>
#include <math.h>

typedef __attribute__((ext_vector_type(16))) __bf16 v16bf;
typedef __attribute__((ext_vector_type(8)))  float  v8f;

#define T_    100
#define NPIX  10000
#define BATCH 2
#define FEATC 400
#define H1C   256
#define H2C   128
#define H3C   512
#define NSC   32

__device__ __forceinline__ float sigmoidf_(float x) { return 1.0f / (1.0f + __expf(-x)); }

// k-index pattern within a 32-wide K tile for 16-bit WMMA A/B operands (wave32).
__device__ __forceinline__ int kpat(int lane, int h) {
    return h + ((h < 8) ? ((lane < 16) ? 0 : 8) : ((lane < 16) ? 8 : 16));
}

#define WMMA_BF16(A, Bv, C) \
    __builtin_amdgcn_wmma_f32_16x16x32_bf16(false, (A), false, (Bv), (short)0, (C), false, false)

// ---------------------------------------------------------------------------
// Tiled bf16 WMMA GEMM: out = relu(A*B + bias), out row-major MxN.
// A tiles: At[b*aBS + ((mt*Ktiles+kt)*32 + lane)*16 + h]
// B tiles: Bt[b*bBS + ((kt*Ntiles+nt)*32 + lane)*16 + h]
// One wave: 2 M-tiles x 4 N-tiles (8 accumulators), software-pipelined loads.
// blockDim = (32,4): wave y covers M-tile pair (blockIdx.y*4+ty).
// ---------------------------------------------------------------------------
__global__ void wmma_gemm_relu(const __bf16* __restrict__ At, const __bf16* __restrict__ Bt,
                               float* __restrict__ out, const float* __restrict__ bias,
                               int Ktiles, int Ntiles, int N,
                               long aBS, long bBS, long oBS)
{
    const int lane = threadIdx.x;
    const int mt0  = (blockIdx.y * blockDim.y + threadIdx.y) * 2;
    const int ntg  = blockIdx.x;
    const int b    = blockIdx.z;

    const __bf16* A0 = At + (size_t)b * aBS + (size_t)mt0 * Ktiles * 512 + lane * 16;
    const __bf16* A1 = A0 + (size_t)Ktiles * 512;
    const __bf16* Bb = Bt + (size_t)b * bBS + lane * 16;

    int  nt[4];
    bool valid[4];
#pragma unroll
    for (int u = 0; u < 4; ++u) {
        int n_   = ntg * 4 + u;
        valid[u] = (n_ < Ntiles);
        nt[u]    = valid[u] ? n_ : 0;   // clamp: load valid memory, discard store
    }

    v8f c00 = {}, c01 = {}, c02 = {}, c03 = {};
    v8f c10 = {}, c11 = {}, c12 = {}, c13 = {};

    // prologue: load kt=0 tiles
    v16bf a0 = *(const v16bf*)A0;
    v16bf a1 = *(const v16bf*)A1;
    v16bf b0 = *(const v16bf*)(Bb + (size_t)nt[0] * 512);
    v16bf b1 = *(const v16bf*)(Bb + (size_t)nt[1] * 512);
    v16bf b2 = *(const v16bf*)(Bb + (size_t)nt[2] * 512);
    v16bf b3 = *(const v16bf*)(Bb + (size_t)nt[3] * 512);

    for (int kt = 0; kt < Ktiles; ++kt) {
        // prefetch kt+1 (clamped) while kt's WMMAs execute
        const int ktn = (kt + 1 < Ktiles) ? kt + 1 : kt;
        const __bf16* An = A0 + (size_t)ktn * 512;
        const __bf16* Bn = Bb + (size_t)ktn * Ntiles * 512;
        v16bf a0n = *(const v16bf*)An;
        v16bf a1n = *(const v16bf*)(An + (size_t)Ktiles * 512);
        v16bf b0n = *(const v16bf*)(Bn + (size_t)nt[0] * 512);
        v16bf b1n = *(const v16bf*)(Bn + (size_t)nt[1] * 512);
        v16bf b2n = *(const v16bf*)(Bn + (size_t)nt[2] * 512);
        v16bf b3n = *(const v16bf*)(Bn + (size_t)nt[3] * 512);

        c00 = WMMA_BF16(a0, b0, c00);
        c10 = WMMA_BF16(a1, b0, c10);
        c01 = WMMA_BF16(a0, b1, c01);
        c11 = WMMA_BF16(a1, b1, c11);
        c02 = WMMA_BF16(a0, b2, c02);
        c12 = WMMA_BF16(a1, b2, c12);
        c03 = WMMA_BF16(a0, b3, c03);
        c13 = WMMA_BF16(a1, b3, c13);

        a0 = a0n; a1 = a1n;
        b0 = b0n; b1 = b1n; b2 = b2n; b3 = b3n;
    }

    // D layout: VGPR r -> M = mt*16 + r + 8*(lane>=16), N = nt*16 + (lane&15)
    const int col = lane & 15;
    float* ob = out + (size_t)b * oBS;
    v8f accs[2][4] = {{c00, c01, c02, c03}, {c10, c11, c12, c13}};
#pragma unroll
    for (int mm = 0; mm < 2; ++mm) {
        const int mrow0 = (mt0 + mm) * 16 + ((lane >> 4) << 3);
#pragma unroll
        for (int u = 0; u < 4; ++u) {
            if (!valid[u]) continue;
            int n = nt[u] * 16 + col;
#pragma unroll
            for (int r = 0; r < 8; ++r) {
                int   m = mrow0 + r;
                float v = accs[mm][u][r] + bias[m];
                ob[(size_t)m * N + n] = fmaxf(v, 0.0f);
            }
        }
    }
}

// ---- operand swizzle kernels (fp32 -> bf16 in WMMA register layout) -------

// A: src row-major MxK (single matrix, e.g. conv weights)
__global__ void tile_a_bf16(const float* __restrict__ src, __bf16* __restrict__ dst,
                            int Mtiles, int Ktiles, int K)
{
    int idx = blockIdx.x * blockDim.x + threadIdx.x;
    int total = Mtiles * Ktiles * 512;
    if (idx >= total) return;
    int h = idx & 15, lane = (idx >> 4) & 31;
    int tile = idx >> 9;
    int kt = tile % Ktiles, mt = tile / Ktiles;
    int m = mt * 16 + (lane & 15);
    int k = kt * 32 + kpat(lane, h);
    dst[idx] = (__bf16)src[(size_t)m * K + k];
}

// B from sample_mask: logical B[k=(d*100+t)][n], stored mask[t*320000 + d*10000 + n]
__global__ void tile_b_mask(const float* __restrict__ mask, __bf16* __restrict__ dst)
{
    int idx = blockIdx.x * blockDim.x + threadIdx.x;
    int total = 100 * 625 * 512;               // Ktiles=100, Ntiles=625
    if (idx >= total) return;
    int h = idx & 15, lane = (idx >> 4) & 31;
    int tile = idx >> 9;
    int nt = tile % 625, kt = tile / 625;
    int k = kt * 32 + kpat(lane, h);
    int n = nt * 16 + (lane & 15);
    int d = k / 100, t = k - d * 100;
    dst[idx] = (__bf16)mask[(size_t)t * 320000 + (size_t)d * 10000 + n];
}

// B from plain row-major KxN (batched)
__global__ void tile_b_rm(const float* __restrict__ src, __bf16* __restrict__ dst,
                          int Ktiles, int Ntiles, int N, long srcBS)
{
    int idx = blockIdx.x * blockDim.x + threadIdx.x;
    int total = BATCH * Ktiles * Ntiles * 512;
    if (idx >= total) return;
    int h = idx & 15, lane = (idx >> 4) & 31;
    int tile = idx >> 9;
    int nt = tile % Ntiles; tile /= Ntiles;
    int kt = tile % Ktiles; int b = tile / Ktiles;
    int k = kt * 32 + kpat(lane, h);
    int n = nt * 16 + (lane & 15);
    dst[idx] = (__bf16)src[(size_t)b * srcBS + (size_t)k * N + n];
}

// B = im2col of (128,100,100) with 3x3, pad 1: K=1152 (c*9 + (di*3+dj)), Ktiles=36
__global__ void tile_b_im2col(const float* __restrict__ src, __bf16* __restrict__ dst, long srcBS)
{
    int idx = blockIdx.x * blockDim.x + threadIdx.x;
    int total = BATCH * 36 * 625 * 512;
    if (idx >= total) return;
    int h = idx & 15, lane = (idx >> 4) & 31;
    int tile = idx >> 9;
    int nt = tile % 625; tile /= 625;
    int kt = tile % 36;  int b = tile / 36;
    int k = kt * 32 + kpat(lane, h);
    int n = nt * 16 + (lane & 15);
    int c = k / 9, r = k - c * 9;
    int i = n / 100 + (r / 3) - 1;
    int j = n % 100 + (r % 3) - 1;
    float v = 0.0f;
    if ((unsigned)i < 100u && (unsigned)j < 100u)
        v = src[(size_t)b * srcBS + (size_t)c * 10000 + i * 100 + j];
    dst[idx] = (__bf16)v;
}

// G[b,o,(d,t)] = sum_c p3d_w[o,c,d] * cm[b,c,t], written directly in A-tile layout
__global__ void gfuse_kernel(const float* __restrict__ p3d_w, const float* __restrict__ cm,
                             __bf16* __restrict__ GA)
{
    int idx = blockIdx.x * blockDim.x + threadIdx.x;
    int total = BATCH * 32 * 100 * 512;        // Mtiles=32, Ktiles=100
    if (idx >= total) return;
    int h = idx & 15, lane = (idx >> 4) & 31;
    int tile = idx >> 9;
    int kt = tile % 100; tile /= 100;
    int mt = tile % 32;  int b = tile / 32;
    int m = mt * 16 + (lane & 15);
    int k = kt * 32 + kpat(lane, h);
    int d = k / 100, t = k - d * 100;
    const float* wp = p3d_w + (size_t)m * H1C * NSC + d;   // stride 32 over c
    const float* cp = cm + (size_t)b * H1C * T_ + t;       // stride 100 over c
    float s = 0.0f;
    for (int c = 0; c < H1C; ++c) s += wp[(size_t)c * NSC] * cp[(size_t)c * T_];
    GA[idx] = (__bf16)s;
}

// ---- small conv / head kernels (negligible FLOPs, plain VALU) -------------

__global__ void conv1d_g_relu(const float* __restrict__ in, const float* __restrict__ w,
                              const float* __restrict__ bias, float* __restrict__ out,
                              int Cin, int Cout, int groups)
{
    int idx = blockIdx.x * blockDim.x + threadIdx.x;
    int total = BATCH * Cout * T_;
    if (idx >= total) return;
    int t = idx % T_;
    int o = (idx / T_) % Cout;
    int b = idx / (T_ * Cout);
    int cpg = Cin / groups, opg = Cout / groups;
    int icb = (o / opg) * cpg;
    float s = bias[o];
    for (int ic = 0; ic < cpg; ++ic) {
        const float* ip = in + ((size_t)b * Cin + icb + ic) * T_;
        const float* wp = w + ((size_t)o * cpg + ic) * 3;
        if (t > 0)      s += wp[0] * ip[t - 1];
        s += wp[1] * ip[t];
        if (t < T_ - 1) s += wp[2] * ip[t + 1];
    }
    out[idx] = fmaxf(s, 0.0f);
}

__global__ void head_sigmoid(const float* __restrict__ in, const float* __restrict__ w,
                             const float* __restrict__ bias, float* __restrict__ out)
{
    int idx = blockIdx.x * blockDim.x + threadIdx.x;
    if (idx >= BATCH * T_) return;
    int b = idx / T_, t = idx % T_;
    float s = bias[0];
    for (int c = 0; c < H1C; ++c) s += w[c] * in[((size_t)b * H1C + c) * T_ + t];
    out[idx] = sigmoidf_(s);
}

__global__ void q4_sigmoid(const float* __restrict__ in, const float* __restrict__ w,
                           const float* __restrict__ bias, float* __restrict__ out)
{
    int idx = blockIdx.x * blockDim.x + threadIdx.x;
    if (idx >= BATCH * 2 * NPIX) return;
    int n = idx % NPIX;
    int m = (idx / NPIX) % 2;
    int b = idx / (2 * NPIX);
    float s = bias[m];
    for (int c = 0; c < H2C; ++c)
        s += w[m * H2C + c] * in[((size_t)b * H2C + c) * NPIX + n];
    out[idx] = sigmoidf_(s);   // conf layout (B,2,T,T) == idx
}

// ---------------------------------------------------------------------------
extern "C" void kernel_launch(void* const* d_in, const int* in_sizes, int n_in,
                              void* d_out, int out_size, void* d_ws, size_t ws_size,
                              hipStream_t stream)
{
    (void)in_sizes; (void)n_in; (void)out_size; (void)ws_size;
    const float* x     = (const float*)d_in[0];
    const float* mask  = (const float*)d_in[1];
    const float* b1_w  = (const float*)d_in[2];  const float* b1_b = (const float*)d_in[3];
    const float* b2_w  = (const float*)d_in[4];  const float* b2_b = (const float*)d_in[5];
    const float* s1_w  = (const float*)d_in[6];  const float* s1_b = (const float*)d_in[7];
    const float* s2_w  = (const float*)d_in[8];  const float* s2_b = (const float*)d_in[9];
    const float* e1_w  = (const float*)d_in[10]; const float* e1_b = (const float*)d_in[11];
    const float* e2_w  = (const float*)d_in[12]; const float* e2_b = (const float*)d_in[13];
    const float* p1_w  = (const float*)d_in[14]; const float* p1_b = (const float*)d_in[15];
    const float* p3d_w = (const float*)d_in[16]; const float* p3d_b = (const float*)d_in[17];
    const float* q1_w  = (const float*)d_in[18]; const float* q1_b = (const float*)d_in[19];
    const float* q2_w  = (const float*)d_in[20]; const float* q2_b = (const float*)d_in[21];
    const float* q3_w  = (const float*)d_in[22]; const float* q3_b = (const float*)d_in[23];
    const float* q4_w  = (const float*)d_in[24]; const float* q4_b = (const float*)d_in[25];
    float* outF = (float*)d_out;

    char* ws = (char*)d_ws;
    size_t off = 0;
    auto alloc = [&](size_t bytes) -> char* {
        char* p = ws + off;
        off = (off + bytes + 255) & ~(size_t)255;
        return p;
    };

    float*  base1 = (float*)alloc((size_t)BATCH * H1C * T_ * 4);
    float*  base  = (float*)alloc((size_t)BATCH * H1C * T_ * 4);
    float*  tmp   = (float*)alloc((size_t)BATCH * H1C * T_ * 4);
    float*  cm    = (float*)alloc((size_t)BATCH * H1C * T_ * 4);
    __bf16* GA    = (__bf16*)alloc((size_t)BATCH * 32 * 100 * 512 * 2);
    __bf16* maskB = (__bf16*)alloc((size_t)100 * 625 * 512 * 2);        // 64 MB
    float*  y     = (float*)alloc((size_t)BATCH * H3C * NPIX * 4);      // 41 MB
    __bf16* yB    = (__bf16*)alloc((size_t)BATCH * 16 * 625 * 512 * 2); // 20 MB
    __bf16* q1A   = (__bf16*)alloc((size_t)8 * 16 * 512 * 2);
    float*  y2    = (float*)alloc((size_t)BATCH * H2C * NPIX * 4);
    __bf16* colB  = (__bf16*)alloc((size_t)BATCH * 36 * 625 * 512 * 2); // 46 MB
    __bf16* q2A   = (__bf16*)alloc((size_t)8 * 36 * 512 * 2);
    __bf16* q3A   = (__bf16*)alloc((size_t)8 * 36 * 512 * 2);
    float*  y3    = (float*)alloc((size_t)BATCH * H2C * NPIX * 4);
    float*  y4    = y;  // y free after yB swizzle; reuse for q3 output

    auto g1 = [](long n) { return dim3((unsigned)((n + 255) / 256)); };

    // backbone + heads
    conv1d_g_relu<<<g1((long)BATCH * H1C * T_), 256, 0, stream>>>(x, b1_w, b1_b, base1, FEATC, H1C, 4);
    conv1d_g_relu<<<g1((long)BATCH * H1C * T_), 256, 0, stream>>>(base1, b2_w, b2_b, base, H1C, H1C, 4);
    conv1d_g_relu<<<g1((long)BATCH * H1C * T_), 256, 0, stream>>>(base, s1_w, s1_b, tmp, H1C, H1C, 4);
    head_sigmoid<<<g1(BATCH * T_), 256, 0, stream>>>(tmp, s2_w, s2_b, outF + 40000);
    conv1d_g_relu<<<g1((long)BATCH * H1C * T_), 256, 0, stream>>>(base, e1_w, e1_b, tmp, H1C, H1C, 4);
    head_sigmoid<<<g1(BATCH * T_), 256, 0, stream>>>(tmp, e2_w, e2_b, outF + 40200);
    conv1d_g_relu<<<g1((long)BATCH * H1C * T_), 256, 0, stream>>>(base, p1_w, p1_b, cm, H1C, H1C, 1);

    // fused p3d: G = W(o,c,d) x cm(c,t), then y = relu(G x mask' + bias)
    gfuse_kernel<<<g1((long)BATCH * 32 * 100 * 512), 256, 0, stream>>>(p3d_w, cm, GA);
    tile_b_mask<<<g1((long)100 * 625 * 512), 256, 0, stream>>>(mask, maskB);
    wmma_gemm_relu<<<dim3(157, 4, 2), dim3(32, 4), 0, stream>>>(
        GA, maskB, y, p3d_b, 100, 625, NPIX,
        (long)32 * 100 * 512, 0L, (long)H3C * NPIX);

    // q1: 1x1 conv 512->128
    tile_b_rm<<<g1((long)BATCH * 16 * 625 * 512), 256, 0, stream>>>(y, yB, 16, 625, NPIX, (long)H3C * NPIX);
    tile_a_bf16<<<g1((long)8 * 16 * 512), 256, 0, stream>>>(q1_w, q1A, 8, 16, 512);
    wmma_gemm_relu<<<dim3(157, 1, 2), dim3(32, 4), 0, stream>>>(
        q1A, yB, y2, q1_b, 16, 625, NPIX,
        0L, (long)16 * 625 * 512, (long)H2C * NPIX);

    // q2: 3x3 conv 128->128 (implicit GEMM)
    tile_b_im2col<<<g1((long)BATCH * 36 * 625 * 512), 256, 0, stream>>>(y2, colB, (long)H2C * NPIX);
    tile_a_bf16<<<g1((long)8 * 36 * 512), 256, 0, stream>>>(q2_w, q2A, 8, 36, 1152);
    wmma_gemm_relu<<<dim3(157, 1, 2), dim3(32, 4), 0, stream>>>(
        q2A, colB, y3, q2_b, 36, 625, NPIX,
        0L, (long)36 * 625 * 512, (long)H2C * NPIX);

    // q3: 3x3 conv 128->128
    tile_b_im2col<<<g1((long)BATCH * 36 * 625 * 512), 256, 0, stream>>>(y3, colB, (long)H2C * NPIX);
    tile_a_bf16<<<g1((long)8 * 36 * 512), 256, 0, stream>>>(q3_w, q3A, 8, 36, 1152);
    wmma_gemm_relu<<<dim3(157, 1, 2), dim3(32, 4), 0, stream>>>(
        q3A, colB, y4, q3_b, 36, 625, NPIX,
        0L, (long)36 * 625 * 512, (long)H2C * NPIX);

    // q4: 1x1 conv 128->2 + sigmoid -> conf
    q4_sigmoid<<<g1((long)BATCH * 2 * NPIX), 256, 0, stream>>>(y4, q4_w, q4_b, outF);
}
